// TransformerSenderGS_2886218023716
// MI455X (gfx1250) — compile-verified, matrix operand order checked by reading, not run on hardware
//
#include <hip/hip_runtime.h>

// ---------------------------------------------------------------------------
// TransformerSenderGS for MI455X (gfx1250): incremental decode w/ KV caches.
// GEMMs: v_wmma_f32_16x16x32_bf16 with double-buffered LDS staging via
// global_load_async_to_lds_b128 (ASYNCcnt pipeline, no VGPR round trip).
// ---------------------------------------------------------------------------

#define D_      512
#define H_      8
#define DH_     64
#define FF_     2048
#define V_      4096
#define NL_     4
#define MAXLEN_ 16
#define B_      128
#define LSRC_   32

#define LDP_    40   // padded LDS row stride in halves (80B = 20 banks)

typedef __attribute__((ext_vector_type(16))) __bf16 v16bf;
typedef __attribute__((ext_vector_type(8)))  float  v8f;

// ---------------- bf16 helpers (bit-level, no host __bf16 use) -------------
__device__ inline unsigned short f2bits(float f) {
  union { float f; unsigned u; } x; x.f = f;
  unsigned u = x.u;
  unsigned r = u + 0x7FFFu + ((u >> 16) & 1u);   // round-to-nearest-even
  return (unsigned short)(r >> 16);
}
__device__ inline __bf16 bfbits(unsigned short h) {
  __bf16 b; __builtin_memcpy(&b, &h, 2); return b;
}
__device__ inline v8f zero8() {
  v8f z;
#pragma unroll
  for (int i = 0; i < 8; ++i) z[i] = 0.0f;
  return z;
}

// 16x32 bf16 fragment loader (A layout; B uses same indexing on W(NxK) rows).
// lane&15 selects row, lane>>4 selects the K+8 / K+24 half-chunks.
__device__ inline v16bf load_frag(const unsigned short* __restrict__ p0, int ld, int lane) {
  const int r = lane & 15, hi = lane >> 4;
  const unsigned short* p = p0 + (size_t)r * ld + hi * 8;
  v16bf f;
#pragma unroll
  for (int j = 0; j < 8; ++j) {
    const int kk = ((j >> 2) << 4) + ((j & 3) << 1);  // 0,2,4,6,16,18,20,22
    f[2 * j]     = bfbits(p[kk]);
    f[2 * j + 1] = bfbits(p[kk + 1]);
  }
  return f;
}

__device__ inline void store_tile(float* __restrict__ C, unsigned short* __restrict__ Cb,
                                  int N, int m0, int n0, const v8f& acc,
                                  const float* __restrict__ bias, int act, int lane) {
  const int r = lane & 15, hi = lane >> 4;
  const int n = n0 + r;
  const float bv = bias ? bias[n] : 0.0f;
#pragma unroll
  for (int q = 0; q < 8; ++q) {
    float v = acc[q] + bv;
    if (act == 1) v = 0.5f * v * (1.0f + erff(v * 0.70710678118654752f));  // exact GELU
    const size_t o = (size_t)(m0 + q + hi * 8) * N + n;
    if (C)  C[o]  = v;
    if (Cb) Cb[o] = f2bits(v);
  }
}

// async memory -> LDS copy of one 16B slot (no VGPR round trip, ASYNCcnt)
__device__ inline void async_cp16(void* lds_dst, const void* gsrc) {
  unsigned dst = (unsigned)(size_t)lds_dst;   // generic LDS ptr low 32b = LDS offset
  asm volatile("global_load_async_to_lds_b128 %0, %1, off"
               :: "v"(dst), "v"(gsrc) : "memory");
}
__device__ inline void wait_async0() {
  asm volatile("s_wait_asynccnt 0x0" ::: "memory");
}

// ---------------- WMMA GEMM: C[MxN] = A[MxK] * W[NxK]^T + bias -------------
// Block = 128 threads = 4 waves. Block tile M=16, N=128; wave tile 16x32
// (2 accumulators). K-chunks (32) double-buffered through LDS with async
// copies overlapping WMMA. Requires M%16==0, N%128==0, K%32==0.
__launch_bounds__(128, 1)
__global__ void k_gemm(const unsigned short* __restrict__ A,
                       const unsigned short* __restrict__ W,
                       const float* __restrict__ bias,
                       float* __restrict__ C, unsigned short* __restrict__ Cb,
                       int M, int N, int K, int act) {
  __shared__ __align__(16) unsigned short As[2][16 * LDP_];
  __shared__ __align__(16) unsigned short Ws[2][128 * LDP_];
  const int tid  = threadIdx.x;
  const int lane = tid & 31;
  const int wave = tid >> 5;              // 0..3
  const int m0 = blockIdx.y * 16;
  const int nb = blockIdx.x * 128;        // block N base
  const int n0 = nb + wave * 32;          // wave N base

  const unsigned short* Ag = A + (size_t)m0 * K;
  const unsigned short* Wg = W + (size_t)nb * K;

  const int r0 = tid >> 1, h0 = tid & 1;            // slots 0..255 (W) / 0..31 (A)
  const int r1 = (tid + 128) >> 1;                  // W slots 128..255

  auto stage = [&](int buf, int k0) {
    if (tid < 32)
      async_cp16(&As[buf][r0 * LDP_ + h0 * 16], Ag + (size_t)r0 * K + k0 + h0 * 16);
    async_cp16(&Ws[buf][r0 * LDP_ + h0 * 16], Wg + (size_t)r0 * K + k0 + h0 * 16);
    async_cp16(&Ws[buf][r1 * LDP_ + h0 * 16], Wg + (size_t)r1 * K + k0 + h0 * 16);
  };

  v8f acc0 = zero8(), acc1 = zero8();

  stage(0, 0);
  wait_async0();
  __syncthreads();

  int buf = 0;
  for (int k0 = 0; k0 < K; k0 += 32) {
    if (k0 + 32 < K) stage(buf ^ 1, k0 + 32);   // overlap copy(k+1) w/ compute(k)
    v16bf a  = load_frag(As[buf], LDP_, lane);
    v16bf b0 = load_frag(Ws[buf] + (wave * 32 +  0) * LDP_, LDP_, lane);
    v16bf b1 = load_frag(Ws[buf] + (wave * 32 + 16) * LDP_, LDP_, lane);
    acc0 = __builtin_amdgcn_wmma_f32_16x16x32_bf16(false, a, false, b0, (short)0, acc0, false, false);
    acc1 = __builtin_amdgcn_wmma_f32_16x16x32_bf16(false, a, false, b1, (short)0, acc1, false, false);
    wait_async0();       // next buffer fully landed in LDS
    __syncthreads();     // all waves done reading current buffer
    buf ^= 1;
  }
  store_tile(C, Cb, N, m0, n0 +  0, acc0, bias, act, lane);
  store_tile(C, Cb, N, m0, n0 + 16, acc1, bias, act, lane);
}

// ---------------- elementwise f32 -> bf16 ----------------------------------
__global__ void k_f32_to_bf16(const float* __restrict__ in,
                              unsigned short* __restrict__ out, long n) {
  long i = (long)blockIdx.x * blockDim.x + threadIdx.x;
  const long stride = (long)gridDim.x * blockDim.x;
  for (; i < n; i += stride) out[i] = f2bits(in[i]);
}

// ---------------- positional embedding table pe[16][512] -------------------
__global__ void k_pos_emb(float* __restrict__ pe) {
  const int i = blockIdx.x * blockDim.x + threadIdx.x;
  if (i >= MAXLEN_ * D_) return;
  const int p = i >> 9, d = i & 511;
  const int a = (d < 256) ? d : d - 256;
  const float denom = __powf(10000.0f, (float)a * (1.0f / 256.0f));
  const float ph = (float)p / denom;
  pe[i] = (d < 256) ? __sinf(ph) : __cosf(ph);
}

// ---------------- step input: x = (special | embed[tok]*sqrt(D)) + pe[s] ---
__global__ void k_step_input(const float* __restrict__ special,
                             const float* __restrict__ embed,
                             const int* __restrict__ tok,
                             const float* __restrict__ pe, int step,
                             float* __restrict__ x,
                             unsigned short* __restrict__ xb) {
  const int b = blockIdx.x, t = threadIdx.x;
  for (int d = t; d < D_; d += 256) {
    float base = (step == 0) ? special[d]
                             : embed[(size_t)tok[b] * D_ + d] * 22.62741699796952f;
    const float v = base + pe[step * D_ + d];
    x[(size_t)b * D_ + d] = v;
    xb[(size_t)b * D_ + d] = f2bits(v);
  }
}

// ---------------- incremental self-attention (block = one (b,h)) -----------
__global__ void k_self_attn(const float* __restrict__ qkv,
                            float* __restrict__ Kc, float* __restrict__ Vc,
                            unsigned short* __restrict__ ctxb, int step) {
  __shared__ float sc[MAXLEN_];
  const int b = blockIdx.x >> 3, hh = blockIdx.x & 7, t = threadIdx.x;  // 64 thr
  const int hd = hh * DH_;
  const float* qk = qkv + (size_t)b * (3 * D_);
  // append current K,V to the cache
  Kc[((size_t)step * B_ + b) * D_ + hd + t] = qk[D_ + hd + t];
  Vc[((size_t)step * B_ + b) * D_ + hd + t] = qk[2 * D_ + hd + t];
  __syncthreads();
  if (t <= step) {
    float s = 0.0f;
    const float* kr = Kc + ((size_t)t * B_ + b) * D_ + hd;
    for (int d = 0; d < DH_; ++d) s += qk[hd + d] * kr[d];
    sc[t] = s * 0.125f;  // 1/sqrt(64)
  }
  __syncthreads();
  const int L = step + 1;
  float m = -1e30f;
  for (int i = 0; i < L; ++i) m = fmaxf(m, sc[i]);
  float denom = 0.0f, o = 0.0f;
  for (int i = 0; i < L; ++i) {
    const float w = __expf(sc[i] - m);
    denom += w;
    o += w * Vc[((size_t)i * B_ + b) * D_ + hd + t];
  }
  ctxb[(size_t)b * D_ + hd + t] = f2bits(o / denom);
}

// ---------------- cross-attention vs precomputed memory K/V ----------------
__global__ void k_cross_attn(const float* __restrict__ q,
                             const float* __restrict__ mkv,  // (LSRC*B) x 1024
                             const unsigned char* __restrict__ mask,
                             unsigned short* __restrict__ ctxb) {
  __shared__ float sc[LSRC_];
  const int b = blockIdx.x >> 3, hh = blockIdx.x & 7, t = threadIdx.x;  // 64 thr
  const int hd = hh * DH_;
  const float* qr = q + (size_t)b * D_ + hd;
  if (t < LSRC_) {
    float s = 0.0f;
    const float* kr = mkv + ((size_t)t * B_ + b) * (2 * D_) + hd;
    for (int d = 0; d < DH_; ++d) s += qr[d] * kr[d];
    s *= 0.125f;
    if (mask[b * LSRC_ + t]) s = -1e9f;
    sc[t] = s;
  }
  __syncthreads();
  float m = -1e30f;
  for (int i = 0; i < LSRC_; ++i) m = fmaxf(m, sc[i]);
  float denom = 0.0f, o = 0.0f;
  for (int i = 0; i < LSRC_; ++i) {
    const float w = __expf(sc[i] - m);
    denom += w;
    o += w * mkv[((size_t)i * B_ + b) * (2 * D_) + D_ + hd + t];
  }
  ctxb[(size_t)b * D_ + hd + t] = f2bits(o / denom);
}

// ---------------- residual add + LayerNorm (block = one row) ---------------
__global__ void k_add_ln(const float* __restrict__ xin, const float* __restrict__ h,
                         const float* __restrict__ g, const float* __restrict__ bta,
                         float* __restrict__ xout, unsigned short* __restrict__ xb) {
  __shared__ float red[256];
  const int bi = blockIdx.x, t = threadIdx.x;
  const float* xr = xin + (size_t)bi * D_;
  const float* hr = h + (size_t)bi * D_;
  const float s0 = xr[t] + hr[t];
  const float s1 = xr[t + 256] + hr[t + 256];
  red[t] = s0 + s1; __syncthreads();
  for (int off = 128; off > 0; off >>= 1) { if (t < off) red[t] += red[t + off]; __syncthreads(); }
  const float mean = red[0] * (1.0f / (float)D_);
  __syncthreads();
  const float d0 = s0 - mean, d1 = s1 - mean;
  red[t] = d0 * d0 + d1 * d1; __syncthreads();
  for (int off = 128; off > 0; off >>= 1) { if (t < off) red[t] += red[t + off]; __syncthreads(); }
  const float inv = rsqrtf(red[0] * (1.0f / (float)D_) + 1e-5f);
  const float o0 = d0 * inv * g[t] + bta[t];
  const float o1 = d1 * inv * g[t + 256] + bta[t + 256];
  xout[(size_t)bi * D_ + t] = o0;       xout[(size_t)bi * D_ + t + 256] = o1;
  xb[(size_t)bi * D_ + t] = f2bits(o0); xb[(size_t)bi * D_ + t + 256] = f2bits(o1);
}

// ---------------- gumbel + softmax + argmax + one-hot + token --------------
__global__ void k_sample(const float* __restrict__ logits, const float* __restrict__ gum,
                         int step, float* __restrict__ seq, float* __restrict__ dist,
                         int* __restrict__ tok, float* __restrict__ tbuf) {
  __shared__ float smax[256]; __shared__ int sidx[256]; __shared__ float ssum[256];
  const int b = blockIdx.x, t = threadIdx.x;
  const float* lr = logits + (size_t)b * V_;
  const float* gr = gum + ((size_t)step * B_ + b) * V_;
  float* tr = tbuf + (size_t)b * V_;
  float lm = -1e30f; int li = 0;
  for (int v = t; v < V_; v += 256) {
    const float g = -logf(-logf(gr[v]));
    const float val = lr[v] + g;  // TEMP == 1
    tr[v] = val;
    if (val > lm) { lm = val; li = v; }
  }
  smax[t] = lm; sidx[t] = li; __syncthreads();
  for (int off = 128; off > 0; off >>= 1) {
    if (t < off) {
      if (smax[t + off] > smax[t] ||
          (smax[t + off] == smax[t] && sidx[t + off] < sidx[t])) {
        smax[t] = smax[t + off]; sidx[t] = sidx[t + off];
      }
    }
    __syncthreads();
  }
  const float M = smax[0]; const int A = sidx[0];
  __syncthreads();
  float ls = 0.0f;
  for (int v = t; v < V_; v += 256) ls += __expf(tr[v] - M);
  ssum[t] = ls; __syncthreads();
  for (int off = 128; off > 0; off >>= 1) { if (t < off) ssum[t] += ssum[t + off]; __syncthreads(); }
  const float inv = 1.0f / ssum[0];
  float* dr = dist + ((size_t)step * B_ + b) * V_;
  float* sr = seq + ((size_t)step * B_ + b) * V_;
  for (int v = t; v < V_; v += 256) {
    dr[v] = __expf(tr[v] - M) * inv;
    sr[v] = (v == A) ? 1.0f : 0.0f;
  }
  if (t == 0) tok[b] = A;
}

// ---------------------------------------------------------------------------
struct LayerP {
  const float *sa_w_in, *sa_b_in, *sa_w_out, *sa_b_out;
  const float *ca_w_in, *ca_b_in, *ca_w_out, *ca_b_out;
  const float *ff_w1, *ff_b1, *ff_w2, *ff_b2;
  const float *ln1_g, *ln1_b, *ln2_g, *ln2_b, *ln3_g, *ln3_b;
};

static inline void gemm(hipStream_t s, const unsigned short* A, const unsigned short* W,
                        const float* bias, float* C, unsigned short* Cb,
                        int M, int N, int K, int act) {
  dim3 g(N / 128, M / 16);
  k_gemm<<<g, 128, 0, s>>>(A, W, bias, C, Cb, M, N, K, act);
}
static inline void tobf(hipStream_t s, const float* in, unsigned short* out, long n) {
  long blocks = (n + 255) / 256; if (blocks > 8192) blocks = 8192;
  k_f32_to_bf16<<<(int)blocks, 256, 0, s>>>(in, out, n);
}

extern "C" void kernel_launch(void* const* d_in, const int* in_sizes, int n_in,
                              void* d_out, int out_size, void* d_ws, size_t ws_size,
                              hipStream_t stream) {
  auto F = [&](int i) { return (const float*)d_in[i]; };
  const unsigned char* maskp = (const unsigned char*)d_in[1];
  const float *enc = F(0), *gum = nullptr, *vocab_w = nullptr, *vocab_b = nullptr,
              *embed = nullptr, *special = nullptr;
  LayerP lp[NL_];
  const bool sorted = (n_in >= 3 && in_sizes[2] == MAXLEN_ * B_ * V_);
  if (!sorted) {  // insertion-order pytree flatten
    int i = 2;
    for (int l = 0; l < NL_; ++l) {
      LayerP& p = lp[l];
      p.sa_w_in = F(i++); p.sa_b_in = F(i++); p.sa_w_out = F(i++); p.sa_b_out = F(i++);
      p.ca_w_in = F(i++); p.ca_b_in = F(i++); p.ca_w_out = F(i++); p.ca_b_out = F(i++);
      p.ff_w1 = F(i++); p.ff_b1 = F(i++); p.ff_w2 = F(i++); p.ff_b2 = F(i++);
      p.ln1_g = F(i++); p.ln1_b = F(i++); p.ln2_g = F(i++); p.ln2_b = F(i++);
      p.ln3_g = F(i++); p.ln3_b = F(i++);
    }
    vocab_w = F(i++); vocab_b = F(i++); embed = F(i++); special = F(i++); gum = F(i++);
  } else {  // alphabetically-sorted pytree flatten
    gum = F(2); embed = F(3);
    int i = 4;
    for (int l = 0; l < NL_; ++l) {
      LayerP& p = lp[l];
      p.ca_b_in = F(i++); p.ca_b_out = F(i++); p.ca_w_in = F(i++); p.ca_w_out = F(i++);
      p.ff_b1 = F(i++); p.ff_b2 = F(i++); p.ff_w1 = F(i++); p.ff_w2 = F(i++);
      p.ln1_b = F(i++); p.ln1_g = F(i++); p.ln2_b = F(i++); p.ln2_g = F(i++);
      p.ln3_b = F(i++); p.ln3_g = F(i++);
      p.sa_b_in = F(i++); p.sa_b_out = F(i++); p.sa_w_in = F(i++); p.sa_w_out = F(i++);
    }
    special = F(i++); vocab_b = F(i++); vocab_w = F(i++);
  }

  // ----- workspace carve-up --------------------------------------------------
  char* wp = (char*)d_ws;
  auto carve = [&](size_t bytes) -> char* {
    char* r = wp; wp += (bytes + 255) & ~(size_t)255; return r;
  };
  unsigned short *sa_in_b[NL_], *sa_out_b[NL_], *ca_in_b[NL_], *ca_out_b[NL_],
                 *ff1_wb[NL_], *ff2_wb[NL_];
  for (int l = 0; l < NL_; ++l) {
    sa_in_b[l]  = (unsigned short*)carve((size_t)3 * D_ * D_ * 2);
    sa_out_b[l] = (unsigned short*)carve((size_t)D_ * D_ * 2);
    ca_in_b[l]  = (unsigned short*)carve((size_t)3 * D_ * D_ * 2);
    ca_out_b[l] = (unsigned short*)carve((size_t)D_ * D_ * 2);
    ff1_wb[l]   = (unsigned short*)carve((size_t)FF_ * D_ * 2);
    ff2_wb[l]   = (unsigned short*)carve((size_t)D_ * FF_ * 2);
  }
  unsigned short* vocab_wb = (unsigned short*)carve((size_t)V_ * D_ * 2);
  unsigned short* enc_b16  = (unsigned short*)carve((size_t)LSRC_ * B_ * D_ * 2);
  float* pe = (float*)carve((size_t)MAXLEN_ * D_ * 4);
  float* memKV[NL_]; float *KcL[NL_], *VcL[NL_];
  for (int l = 0; l < NL_; ++l) memKV[l] = (float*)carve((size_t)LSRC_ * B_ * 2 * D_ * 4);
  for (int l = 0; l < NL_; ++l) {
    KcL[l] = (float*)carve((size_t)MAXLEN_ * B_ * D_ * 4);
    VcL[l] = (float*)carve((size_t)MAXLEN_ * B_ * D_ * 4);
  }
  float* x            = (float*)carve((size_t)B_ * D_ * 4);
  unsigned short* xb  = (unsigned short*)carve((size_t)B_ * D_ * 2);
  float* qkv          = (float*)carve((size_t)B_ * 3 * D_ * 4);
  unsigned short* cxb = (unsigned short*)carve((size_t)B_ * D_ * 2);
  float* hbuf         = (float*)carve((size_t)B_ * D_ * 4);
  unsigned short* f1b = (unsigned short*)carve((size_t)B_ * FF_ * 2);
  float* logits       = (float*)carve((size_t)B_ * V_ * 4);
  float* tbuf         = (float*)carve((size_t)B_ * V_ * 4);
  int* tok            = (int*)carve((size_t)B_ * 4);
  (void)ws_size; (void)out_size;

  // ----- one-time per launch: bf16 weights, pos-emb, memory K/V --------------
  for (int l = 0; l < NL_; ++l) {
    tobf(stream, lp[l].sa_w_in,  sa_in_b[l],  (long)3 * D_ * D_);
    tobf(stream, lp[l].sa_w_out, sa_out_b[l], (long)D_ * D_);
    tobf(stream, lp[l].ca_w_in,  ca_in_b[l],  (long)3 * D_ * D_);
    tobf(stream, lp[l].ca_w_out, ca_out_b[l], (long)D_ * D_);
    tobf(stream, lp[l].ff_w1,    ff1_wb[l],   (long)FF_ * D_);
    tobf(stream, lp[l].ff_w2,    ff2_wb[l],   (long)D_ * FF_);
  }
  tobf(stream, vocab_w, vocab_wb, (long)V_ * D_);
  tobf(stream, enc, enc_b16, (long)LSRC_ * B_ * D_);
  k_pos_emb<<<(MAXLEN_ * D_ + 255) / 256, 256, 0, stream>>>(pe);
  // cross-attn memory K/V per layer: (LSRC*B, D) @ wk/wv^T  -> (LSRC*B, 2D)
  for (int l = 0; l < NL_; ++l)
    gemm(stream, enc_b16, ca_in_b[l] + (size_t)D_ * D_, lp[l].ca_b_in + D_,
         memKV[l], nullptr, LSRC_ * B_, 2 * D_, D_, 0);

  // ----- autoregressive decode ----------------------------------------------
  float* seq = (float*)d_out;
  float* dist = seq + (size_t)MAXLEN_ * B_ * V_;
  for (int s = 0; s < MAXLEN_; ++s) {
    k_step_input<<<B_, 256, 0, stream>>>(special, embed, tok, pe, s, x, xb);
    for (int l = 0; l < NL_; ++l) {
      // self-attention
      gemm(stream, xb, sa_in_b[l], lp[l].sa_b_in, qkv, nullptr, B_, 3 * D_, D_, 0);
      k_self_attn<<<B_ * H_, DH_, 0, stream>>>(qkv, KcL[l], VcL[l], cxb, s);
      gemm(stream, cxb, sa_out_b[l], lp[l].sa_b_out, hbuf, nullptr, B_, D_, D_, 0);
      k_add_ln<<<B_, 256, 0, stream>>>(x, hbuf, lp[l].ln1_g, lp[l].ln1_b, x, xb);
      // cross-attention (wq = first D rows of ca_w_in)
      gemm(stream, xb, ca_in_b[l], lp[l].ca_b_in, qkv, nullptr, B_, D_, D_, 0);
      k_cross_attn<<<B_ * H_, DH_, 0, stream>>>(qkv, memKV[l], maskp, cxb);
      gemm(stream, cxb, ca_out_b[l], lp[l].ca_b_out, hbuf, nullptr, B_, D_, D_, 0);
      k_add_ln<<<B_, 256, 0, stream>>>(x, hbuf, lp[l].ln2_g, lp[l].ln2_b, x, xb);
      // feed-forward (exact GELU + bf16 output fused into GEMM1 epilogue)
      gemm(stream, xb, ff1_wb[l], lp[l].ff_b1, nullptr, f1b, B_, FF_, D_, 1);
      gemm(stream, f1b, ff2_wb[l], lp[l].ff_b2, hbuf, nullptr, B_, D_, FF_, 0);
      k_add_ln<<<B_, 256, 0, stream>>>(x, hbuf, lp[l].ln3_g, lp[l].ln3_b, x, xb);
    }
    gemm(stream, xb, vocab_wb, vocab_b, logits, nullptr, B_, V_, D_, 0);
    k_sample<<<B_, 256, 0, stream>>>(logits, gum, s, seq, dist, tok, tbuf);
  }
}
